// LinformerAttention_3100966388339
// MI455X (gfx1250) — matrix operand order repaired and checked by reading, not compiled, not run on hardware
//
#include <hip/hip_runtime.h>
#include <hip/hip_bf16.h>

typedef __attribute__((ext_vector_type(16))) _Float16 v16h;
typedef __attribute__((ext_vector_type(8)))  _Float16 v8h;
typedef __attribute__((ext_vector_type(8)))  float    v8f;

#define SEQ_LEN 4096
#define DIM     1024
#define NHEADS  16
#define DH      64
#define KPROJ   256
#define BATCH   4

// Padded LDS strides (f16 elems): dword stride = 4*odd (mod 64) -> spread the
// 4-bank groups hit by ds_load_b128 fragment reads across lanes.
#define SA_LD   40    // A tiles, 32 cols + 8 pad
#define SB_LD   136   // B tiles (128 cols) + 8 pad
#define SBT_LD  72    // B tiles (64 cols) + 8 pad
#define SP_LD   264   // attention P tile (256 cols) + 8 pad

// ---------------------------------------------------------------------------
// gfx1250 async global->LDS (ASYNCcnt) with safe fallback
// Builtin signature (probe-confirmed by diagnostics): pointee is a 16B int
// vector; arg0 in AS1 (__device__), arg1 in AS3 (__shared__).
// ---------------------------------------------------------------------------
#if defined(__gfx1250__) && __has_builtin(__builtin_amdgcn_global_load_async_to_lds_b128)
#define HAVE_ASYNC_LDS 1
typedef int v4i_ __attribute__((vector_size(16)));
typedef __attribute__((address_space(1))) v4i_ as1_v4i;
typedef __attribute__((address_space(3))) v4i_ as3_v4i;
#else
#define HAVE_ASYNC_LDS 0
#endif

__device__ inline void cp_async16(const _Float16* g, _Float16* l) {
#if HAVE_ASYNC_LDS
  __builtin_amdgcn_global_load_async_to_lds_b128((as1_v4i*)g, (as3_v4i*)l, 0, 0);
#else
  *(v8h*)l = *(const v8h*)g;
#endif
}

__device__ inline void wait_async0() {
#if HAVE_ASYNC_LDS
#if __has_builtin(__builtin_amdgcn_s_wait_asynccnt)
  __builtin_amdgcn_s_wait_asynccnt(0);
#else
  asm volatile("s_wait_asynccnt 0x0" ::: "memory");
#endif
#endif
}

// ---------------------------------------------------------------------------
// WMMA fragment helpers (v_wmma_f32_16x16x32_f16, wave32 layouts per ISA 7.12.2)
// ---------------------------------------------------------------------------
// A 16x32 (MxK): lane&15 = row M; lanes<16 hold K 0..7 / 16..23, lanes>=16 hold
// K 8..15 / 24..31 (pairs per VGPR).  p0 points at (row0, k0), ld = elems/row.
__device__ inline v16h load_frag_a(const _Float16* p0, int ld) {
  const int lane = threadIdx.x & 31;
  const int kb = (lane & 16) ? 8 : 0;
  const _Float16* p = p0 + (size_t)(lane & 15) * ld + kb;
  v16h a;
#pragma unroll
  for (int i = 0; i < 8; ++i) { a[i] = p[i]; a[i + 8] = p[16 + i]; }
  return a;
}

// B 32x16 (KxN): lane&15 = col N; lanes<16 hold K 0..7 / 16..23, lanes>=16 hold
// K 8..15 / 24..31 (same K order as A).
__device__ inline v16h load_frag_b(const _Float16* p0, int ld) {
  const int lane = threadIdx.x & 31;
  const int kb = (lane & 16) ? 8 : 0;
  const _Float16* p = p0 + (size_t)kb * ld + (lane & 15);
  v16h b;
#pragma unroll
  for (int i = 0; i < 8; ++i) { b[i] = p[(size_t)i * ld]; b[i + 8] = p[(size_t)(16 + i) * ld]; }
  return b;
}

__device__ inline v8f wmma_f16(v16h a, v16h b, v8f c) {
  return __builtin_amdgcn_wmma_f32_16x16x32_f16(false, a, false, b,
                                                (short)0, c, false, false);
}

// ---------------------------------------------------------------------------
// f32 -> f16 conversion
// ---------------------------------------------------------------------------
__global__ void cvt_f32_to_f16_kernel(const float* __restrict__ x,
                                      _Float16* __restrict__ y, int n) {
  int i = blockIdx.x * blockDim.x + threadIdx.x;
  if (i < n) y[i] = (_Float16)x[i];
}

// ---------------------------------------------------------------------------
// NN GEMM: C[M,N] = A[M,K] @ B[K,N] (+ bias)
// Block tile 128x128, 256 threads (8 waves), wave tile 32x64 (2x4 WMMA frags).
// Double-buffered LDS filled with async global->LDS copies.
// All dims multiples of 128/32 in this problem -> no bounds checks.
// ---------------------------------------------------------------------------
template <bool F32OUT>
__global__ void __launch_bounds__(256)
gemm_nn_bias_kernel(const _Float16* __restrict__ A, int lda,
                    const _Float16* __restrict__ Bm, int ldb,
                    const float* __restrict__ bias,
                    void* __restrict__ Cout, int ldc, int K) {
  __shared__ __attribute__((aligned(16))) _Float16 sA[2][128 * SA_LD];
  __shared__ __attribute__((aligned(16))) _Float16 sB[2][32 * SB_LD];
  const int t = threadIdx.x;
  const int wave = t >> 5, lane = t & 31;
  const int wm = wave >> 1, wn = wave & 1;       // 4x2 wave grid
  const int m0 = blockIdx.y * 128, n0 = blockIdx.x * 128;

  // staging coords: A 128x32 (16B per thread x2), B 32x128 (16B x2)
  const int ar = t >> 2, ac = (t & 3) * 8;
  const int br = t >> 4, bc = (t & 15) * 8;

  auto stage = [&](int buf, int k0) {
    cp_async16(A + (size_t)(m0 + ar) * lda + k0 + ac,       sA[buf] + ar * SA_LD + ac);
    cp_async16(A + (size_t)(m0 + ar + 64) * lda + k0 + ac,  sA[buf] + (ar + 64) * SA_LD + ac);
    cp_async16(Bm + (size_t)(k0 + br) * ldb + n0 + bc,      sB[buf] + br * SB_LD + bc);
    cp_async16(Bm + (size_t)(k0 + br + 16) * ldb + n0 + bc, sB[buf] + (br + 16) * SB_LD + bc);
  };

  v8f acc[2][4] = {};
  stage(0, 0);
  wait_async0();
  __syncthreads();
  for (int k0 = 0; k0 < K; k0 += 32) {
    const int cur = (k0 >> 5) & 1;
    if (k0 + 32 < K) stage(cur ^ 1, k0 + 32);   // overlap next tile with math
    v16h a0 = load_frag_a(sA[cur] + (wm * 32) * SA_LD, SA_LD);
    v16h a1 = load_frag_a(sA[cur] + (wm * 32 + 16) * SA_LD, SA_LD);
#pragma unroll
    for (int nt = 0; nt < 4; ++nt) {
      v16h b = load_frag_b(sB[cur] + wn * 64 + nt * 16, SB_LD);
      acc[0][nt] = wmma_f16(a0, b, acc[0][nt]);
      acc[1][nt] = wmma_f16(a1, b, acc[1][nt]);
    }
    wait_async0();
    __syncthreads();
  }
  const int coln = lane & 15;
#pragma unroll
  for (int mi = 0; mi < 2; ++mi) {
    const int rbase = m0 + wm * 32 + mi * 16 + 8 * (lane >> 4);
#pragma unroll
    for (int nt = 0; nt < 4; ++nt) {
      int col = n0 + wn * 64 + nt * 16 + coln;
      float bv = bias ? bias[col] : 0.f;
      if (F32OUT) {
        float* C = (float*)Cout;
#pragma unroll
        for (int j = 0; j < 8; ++j)
          C[(size_t)(rbase + j) * ldc + col] = acc[mi][nt][j] + bv;
      } else {
        _Float16* C = (_Float16*)Cout;
#pragma unroll
        for (int j = 0; j < 8; ++j)
          C[(size_t)(rbase + j) * ldc + col] = (_Float16)(acc[mi][nt][j] + bv);
      }
    }
  }
}

// ---------------------------------------------------------------------------
// TT GEMM core: C[M,N] = A^T @ B where A is stored [K,M] (lda) and B [K,N]
// (ldb).  Used for kE = k^T E and vF = F^T v.  Tile 64x64, 128 threads.
// ---------------------------------------------------------------------------
__device__ void gemm_tt_64x64(const _Float16* __restrict__ A, int lda,
                              const _Float16* __restrict__ Bm, int ldb,
                              _Float16* __restrict__ C, int ldc, int K) {
  __shared__ __attribute__((aligned(16))) _Float16 sA[64 * SA_LD];
  __shared__ __attribute__((aligned(16))) _Float16 sB[32 * SBT_LD];
  const int t = threadIdx.x;
  const int wave = t >> 5;
  const int lane = t & 31;
  const int m0 = blockIdx.y * 64;
  const int n0 = blockIdx.x * 64;

  v8f acc[4] = {};
  for (int k0 = 0; k0 < K; k0 += 32) {
    // stage A transposed: sA[m][kk] = A[k0+kk][m0+m]; coalesced over m
#pragma unroll
    for (int i = 0; i < 16; ++i) {
      int idx = i * 128 + t;   // 0..2047
      int kk = idx >> 6;       // 0..31
      int r = idx & 63;        // 0..63
      sA[r * SA_LD + kk] = A[(size_t)(k0 + kk) * lda + m0 + r];
    }
    {  // stage B 32x64 (async b128 copies)
      int r = t >> 3, c = (t & 7) * 8;
      cp_async16(Bm + (size_t)(k0 + r) * ldb + n0 + c,      sB + r * SBT_LD + c);
      cp_async16(Bm + (size_t)(k0 + r + 16) * ldb + n0 + c, sB + (r + 16) * SBT_LD + c);
    }
    wait_async0();
    __syncthreads();
    v16h a = load_frag_a(sA + (wave * 16) * SA_LD, SA_LD);
#pragma unroll
    for (int nt = 0; nt < 4; ++nt) {
      v16h b = load_frag_b(sB + nt * 16, SBT_LD);
      acc[nt] = wmma_f16(a, b, acc[nt]);
    }
    __syncthreads();
  }
  const int rbase = m0 + wave * 16 + 8 * (lane >> 4);
  const int coln = lane & 15;
#pragma unroll
  for (int nt = 0; nt < 4; ++nt) {
    int col = n0 + nt * 16 + coln;
#pragma unroll
    for (int j = 0; j < 8; ++j)
      C[(size_t)(rbase + j) * ldc + col] = (_Float16)acc[nt][j];
  }
}

// kE[bh][d][m] = sum_l k[b,l,h*64+d] * E[l,m]   (C = 64 x 256)
__global__ void kproj_kernel(const _Float16* __restrict__ kh,
                             const _Float16* __restrict__ Eh,
                             _Float16* __restrict__ keh) {
  int bh = blockIdx.z, b = bh >> 4, h = bh & 15;
  gemm_tt_64x64(kh + (size_t)b * SEQ_LEN * DIM + h * DH, DIM,
                Eh, KPROJ,
                keh + (size_t)bh * DH * KPROJ, KPROJ, SEQ_LEN);
}

// vF[bh][m][d] = sum_l F[l,m] * v[b,l,h*64+d]   (C = 256 x 64)
__global__ void vproj_kernel(const _Float16* __restrict__ Fh,
                             const _Float16* __restrict__ vh,
                             _Float16* __restrict__ vfh) {
  int bh = blockIdx.z, b = bh >> 4, h = bh & 15;
  gemm_tt_64x64(Fh, KPROJ,
                vh + (size_t)b * SEQ_LEN * DIM + h * DH, DIM,
                vfh + (size_t)bh * KPROJ * DH, DH, SEQ_LEN);
}

// ---------------------------------------------------------------------------
// Fused attention: per block = 64 query rows of one (b,h).
//   S = q(64x64) @ kE(64x256) * scale  -> softmax rows -> P (f16, LDS)
//   O = P(64x256) @ vF(256x64)         -> out[b, l, h*64 + d]  (f16)
// ---------------------------------------------------------------------------
__global__ void attn_kernel(const _Float16* __restrict__ qh,   // [B*L, D]
                            const _Float16* __restrict__ kE,   // [BH,64,256]
                            const _Float16* __restrict__ vF,   // [BH,256,64]
                            _Float16* __restrict__ oh) {       // [B*L, D]
  __shared__ __attribute__((aligned(16))) _Float16 sP[64 * SP_LD];
  const int t = threadIdx.x;
  const int wave = t >> 5;
  const int lane = t & 31;
  const int l0 = blockIdx.x * 64;
  const int bh = blockIdx.y;
  const int b = bh >> 4, h = bh & 15;

  const _Float16* qbase = qh + ((size_t)b * SEQ_LEN + l0) * DIM + h * DH;
  const _Float16* ke = kE + (size_t)bh * DH * KPROJ;
  const _Float16* vf = vF + (size_t)bh * KPROJ * DH;

  // ---- S = q @ kE : 16 N-tiles of 16x16, K = 64 (2 wmma steps per tile)
  v8f acc[16] = {};
#pragma unroll
  for (int ks = 0; ks < 2; ++ks) {
    v16h a = load_frag_a(qbase + (size_t)(wave * 16) * DIM + ks * 32, DIM);
#pragma unroll
    for (int nt = 0; nt < 16; ++nt) {
      v16h bfr = load_frag_b(ke + (size_t)(ks * 32) * KPROJ + nt * 16, KPROJ);
      acc[nt] = wmma_f16(a, bfr, acc[nt]);
    }
  }

  // ---- softmax over 256 cols; lane holds col (lane&15)+16*nt for 8 rows
  const float scale = 0.125f;  // dh^-0.5
#pragma unroll
  for (int j = 0; j < 8; ++j) {
    float mx = -1e30f;
#pragma unroll
    for (int nt = 0; nt < 16; ++nt) {
      float v = acc[nt][j] * scale;
      acc[nt][j] = v;
      mx = fmaxf(mx, v);
    }
#pragma unroll
    for (int off = 1; off < 16; off <<= 1)
      mx = fmaxf(mx, __shfl_xor(mx, off, 32));  // 16-lane col groups
    float sum = 0.f;
#pragma unroll
    for (int nt = 0; nt < 16; ++nt) {
      float e = __expf(acc[nt][j] - mx);
      acc[nt][j] = e;
      sum += e;
    }
#pragma unroll
    for (int off = 1; off < 16; off <<= 1)
      sum += __shfl_xor(sum, off, 32);
    float inv = 1.f / sum;
    int rowp = wave * 16 + j + 8 * (lane >> 4);
    int colp = lane & 15;
#pragma unroll
    for (int nt = 0; nt < 16; ++nt)
      sP[rowp * SP_LD + nt * 16 + colp] = (_Float16)(acc[nt][j] * inv);
  }
  __syncthreads();

  // ---- O = P @ vF : 4 N-tiles, K = 256 (8 wmma steps)
  v8f oacc[4] = {};
#pragma unroll
  for (int ks = 0; ks < 8; ++ks) {
    v16h a = load_frag_a(sP + (wave * 16) * SP_LD + ks * 32, SP_LD);
#pragma unroll
    for (int nt = 0; nt < 4; ++nt) {
      v16h bfr = load_frag_b(vf + (size_t)(ks * 32) * DH + nt * 16, DH);
      oacc[nt] = wmma_f16(a, bfr, oacc[nt]);
    }
  }
  _Float16* ob = oh + ((size_t)b * SEQ_LEN + l0) * DIM + h * DH;
  const int rbase = wave * 16 + 8 * (lane >> 4);
  const int coln = lane & 15;
#pragma unroll
  for (int nt = 0; nt < 4; ++nt)
#pragma unroll
    for (int j = 0; j < 8; ++j)
      ob[(size_t)(rbase + j) * DIM + nt * 16 + coln] = (_Float16)oacc[nt][j];
}

// ---------------------------------------------------------------------------
// Launch
// ---------------------------------------------------------------------------
extern "C" void kernel_launch(void* const* d_in, const int* in_sizes, int n_in,
                              void* d_out, int out_size, void* d_ws, size_t ws_size,
                              hipStream_t stream) {
  (void)in_sizes; (void)n_in; (void)out_size; (void)ws_size;
  const float* x  = (const float*)d_in[0];
  const float* Wq = (const float*)d_in[1];
  const float* bq = (const float*)d_in[2];
  const float* Wk = (const float*)d_in[3];
  const float* bk = (const float*)d_in[4];
  const float* Wv = (const float*)d_in[5];
  const float* bv = (const float*)d_in[6];
  const float* E  = (const float*)d_in[7];
  const float* F  = (const float*)d_in[8];
  const float* Wo = (const float*)d_in[9];
  const float* bo = (const float*)d_in[10];
  float* out = (float*)d_out;

  char* ws = (char*)d_ws;
  size_t off = 0;
  auto take = [&](size_t elems) {
    _Float16* p = (_Float16*)(ws + off);
    off += (elems * sizeof(_Float16) + 255) & ~(size_t)255;
    return p;
  };
  const size_t NBL = (size_t)BATCH * SEQ_LEN;       // 16384
  _Float16* xh  = take(NBL * DIM);
  _Float16* wqh = take((size_t)DIM * DIM);
  _Float16* wkh = take((size_t)DIM * DIM);
  _Float16* wvh = take((size_t)DIM * DIM);
  _Float16* woh = take((size_t)DIM * DIM);
  _Float16* Eh  = take((size_t)SEQ_LEN * KPROJ);
  _Float16* Fh  = take((size_t)SEQ_LEN * KPROJ);
  _Float16* qh  = take(NBL * DIM);
  _Float16* kh  = take(NBL * DIM);
  _Float16* vh  = take(NBL * DIM);
  _Float16* keh = take((size_t)BATCH * NHEADS * DH * KPROJ);
  _Float16* vfh = take((size_t)BATCH * NHEADS * KPROJ * DH);
  _Float16* ohb = take(NBL * DIM);

  auto cvt = [&](const float* src, _Float16* dst, int n) {
    cvt_f32_to_f16_kernel<<<(n + 255) / 256, 256, 0, stream>>>(src, dst, n);
  };
  cvt(x,  xh,  (int)(NBL * DIM));
  cvt(Wq, wqh, DIM * DIM);
  cvt(Wk, wkh, DIM * DIM);
  cvt(Wv, wvh, DIM * DIM);
  cvt(Wo, woh, DIM * DIM);
  cvt(E,  Eh,  SEQ_LEN * KPROJ);
  cvt(F,  Fh,  SEQ_LEN * KPROJ);

  dim3 gProj(DIM / 128, (unsigned)(NBL / 128));  // (8, 128)
  gemm_nn_bias_kernel<false><<<gProj, 256, 0, stream>>>(xh, DIM, wqh, DIM, bq, qh, DIM, DIM);
  gemm_nn_bias_kernel<false><<<gProj, 256, 0, stream>>>(xh, DIM, wkh, DIM, bk, kh, DIM, DIM);
  gemm_nn_bias_kernel<false><<<gProj, 256, 0, stream>>>(xh, DIM, wvh, DIM, bv, vh, DIM, DIM);

  kproj_kernel<<<dim3(KPROJ / 64, 1, BATCH * NHEADS), 128, 0, stream>>>(kh, Eh, keh);
  vproj_kernel<<<dim3(1, KPROJ / 64, BATCH * NHEADS), 128, 0, stream>>>(Fh, vh, vfh);

  attn_kernel<<<dim3(SEQ_LEN / 64, BATCH * NHEADS), 128, 0, stream>>>(qh, keh, vfh, ohb);

  gemm_nn_bias_kernel<true><<<gProj, 256, 0, stream>>>(ohb, DIM, woh, DIM, bo, out, DIM, DIM);
}
